// ResnetBlocWithAttn_62354335203904
// MI455X (gfx1250) — matrix operand order repaired and checked by reading, not compiled, not run on hardware
//
#include <hip/hip_runtime.h>
#include <hip/hip_bf16.h>
#include <math.h>

// ---------------------------------------------------------------------------
// Types for CDNA5 WMMA (gfx1250, wave32)
// ---------------------------------------------------------------------------
typedef __bf16 bf16;
typedef bf16  bf16x8  __attribute__((ext_vector_type(8)));
typedef bf16  bf16x16 __attribute__((ext_vector_type(16)));
typedef float f32x8   __attribute__((ext_vector_type(8)));
typedef unsigned int u32x4 __attribute__((ext_vector_type(4)));
typedef int i32x4 __attribute__((ext_vector_type(4)));
typedef int i32x8 __attribute__((ext_vector_type(8)));

#define CCH   256
#define SPT   1024            // 32*32 spatial
#define BATCH 8
#define MTOT  (BATCH*SPT)     // 8192 (b,s) rows
#define KCONV (9*CCH)         // 2304

#if __has_builtin(__builtin_amdgcn_tensor_load_to_lds)
#define HAVE_TDM 1
#else
#define HAVE_TDM 0
#endif

__device__ __forceinline__ f32x8 zf8() {
    f32x8 z;
#pragma unroll
    for (int i = 0; i < 8; ++i) z[i] = 0.f;
    return z;
}

__device__ __forceinline__ bf16x16 zb16() {
    bf16x16 z;
#pragma unroll
    for (int i = 0; i < 16; ++i) z[i] = (bf16)0.f;
    return z;
}

// D = A(16x32 bf16) * B(32x16 bf16) + C(16x16 f32)
__device__ __forceinline__ f32x8 wmma_bf16(bf16x16 a, bf16x16 b, f32x8 c) {
    return __builtin_amdgcn_wmma_f32_16x16x32_bf16(
        /*neg_a=*/false, a, /*neg_b=*/false, b,
        /*c_mod=*/(short)0, c, /*reuse_a=*/false, /*reuse_b=*/false);
}

// A fragment (16x32 bf16). rowPtr -> 32 contiguous bf16 K-values of this
// lane's row (row = lane&15). half = lane>>4.
//   half 0: elements = K[0..15]
//   half 1: elements = K[16..31]
__device__ __forceinline__ bf16x16 load_a_frag(const bf16* rowPtr, int half) {
    return *(const bf16x16*)(rowPtr + 16 * half);
}

// B fragment (32x16 bf16). colPtr -> 32 contiguous bf16 K-values of this
// lane's column (col = lane&15). Lanes 0-15 take K[0..15], 16-31 K[16..31].
__device__ __forceinline__ bf16x16 load_b_frag(const bf16* colPtr, int half) {
    return *(const bf16x16*)(colPtr + 16 * half);
}

// ---------------------------------------------------------------------------
// Tensor Data Mover: issue DMA of a 2-D bf16 panel (tile_h rows x tile_w
// elems, global row stride row_stride elems) into LDS at byte offset lds_off.
// D# layout per CDNA5 ISA sec.8 (group0 128b, group1 256b).
// Issue from ONE wave only (TDM ignores EXEC); TENSORcnt is per-wave, so the
// same wave calls tdm_wait() and the block publishes LDS via __syncthreads().
// ---------------------------------------------------------------------------
__device__ __forceinline__ void tdm_issue_2d(const void* gptr, unsigned lds_off,
                                             unsigned tile_w, unsigned tile_h,
                                             unsigned row_stride)
{
#if HAVE_TDM
    unsigned long long ga = (unsigned long long)gptr;
    u32x4 g0;
    g0[0] = 1u;                                   // count=1, user descriptor
    g0[1] = lds_off;                              // lds_addr (bytes)
    g0[2] = (unsigned)(ga & 0xffffffffu);         // global_addr[31:0]
    g0[3] = (unsigned)((ga >> 32) & 0x01ffffffu)  // global_addr[56:32]
          | (2u << 30);                           // type=2 ("image")
    unsigned td0 = row_stride;                    // tensor_dim0 (elems)
    unsigned td1 = tile_h;                        // tensor_dim1 (rows)
    i32x8 g1;
    g1[0] = 0x00010000;                           // data_size=1 (2 bytes)
    g1[1] = (int)((td0 & 0xffffu) << 16);         // tensor_dim0[15:0]
    g1[2] = (int)((td0 >> 16) | ((td1 & 0xffffu) << 16));
    g1[3] = (int)((td1 >> 16) | (tile_w << 16));  // tile_dim0
    g1[4] = (int)(tile_h & 0xffffu);              // tile_dim1 (tile_dim2=0)
    g1[5] = (int)row_stride;                      // tensor_dim0_stride[31:0]
    g1[6] = 0;                                    // stride hi / dim1_stride lo
    g1[7] = 0;
    i32x4 gz; gz[0] = gz[1] = gz[2] = gz[3] = 0;
#if __clang_major__ >= 23
    i32x8 gz8;
#pragma unroll
    for (int i = 0; i < 8; ++i) gz8[i] = 0;
    __builtin_amdgcn_tensor_load_to_lds(g0, g1, gz, gz, gz8, 0);
#else
    __builtin_amdgcn_tensor_load_to_lds(g0, g1, gz, gz, 0);
#endif
#else
    (void)gptr; (void)lds_off; (void)tile_w; (void)tile_h; (void)row_stride;
#endif
}

__device__ __forceinline__ void tdm_wait()
{
#if HAVE_TDM
    __builtin_amdgcn_s_wait_tensorcnt((short)0);
#endif
}

// ---------------------------------------------------------------------------
// GroupNorm (+optional swish) : f32 in (NCHW or NHWC) -> bf16 NHWC out
// one block per (batch, group); 8 channels x 1024 spatial = 8192 elems
// ---------------------------------------------------------------------------
template <bool IN_NCHW, bool SWISH>
__global__ __launch_bounds__(256) void groupnorm_kernel(
    const float* __restrict__ in, const float* __restrict__ gamma,
    const float* __restrict__ beta, bf16* __restrict__ out)
{
    __shared__ float r1[256];
    __shared__ float r2[256];
    const int tid = threadIdx.x;
    const int b = blockIdx.x >> 5;   // 8 batches
    const int g = blockIdx.x & 31;   // 32 groups

    float sum = 0.f, sq = 0.f;
    for (int e = tid; e < 8192; e += 256) {
        int c = e >> 10, s = e & 1023;
        float v = IN_NCHW
            ? in[(((size_t)b * CCH + g * 8 + c) << 10) + s]
            : in[(((size_t)b << 10) + s) * CCH + g * 8 + c];
        sum += v; sq += v * v;
    }
    r1[tid] = sum; r2[tid] = sq;
    __syncthreads();
#pragma unroll
    for (int o = 128; o > 0; o >>= 1) {
        if (tid < o) { r1[tid] += r1[tid + o]; r2[tid] += r2[tid + o]; }
        __syncthreads();
    }
    const float mean = r1[0] * (1.f / 8192.f);
    const float var  = r2[0] * (1.f / 8192.f) - mean * mean;
    const float rstd = rsqrtf(var + 1e-5f);

    for (int e = tid; e < 8192; e += 256) {
        int c = e >> 10, s = e & 1023;
        int ch = g * 8 + c;
        float v = IN_NCHW
            ? in[(((size_t)b * CCH + ch) << 10) + s]
            : in[(((size_t)b << 10) + s) * CCH + ch];
        float y = (v - mean) * rstd * gamma[ch] + beta[ch];
        if (SWISH) y = y / (1.f + __expf(-y));
        out[(((size_t)b << 10) + s) * CCH + ch] = (bf16)y;
    }
}

// ---------------------------------------------------------------------------
// Implicit-GEMM 3x3 SAME conv: act bf16 NHWC [MTOT][256], weights bf16
// [cout][tap*256+cin] ([256][2304]) -> f32 NHWC out (+bias).
// WG tile: M=128 (spatial) x N=128 (cout); 8 waves, wave tile 32x64.
// Weight panels (128x256 bf16 = 64 KB per tap) are DMA'd by the TDM into a
// DOUBLE-BUFFERED 128 KB LDS region: tap t+1's DMA overlaps tap t's 64 WMMAs.
// ---------------------------------------------------------------------------
__global__ __launch_bounds__(256) void conv3x3_wmma_kernel(
    const bf16* __restrict__ act, const bf16* __restrict__ wt,
    const float* __restrict__ bias, float* __restrict__ out)
{
    __shared__ bf16 ldsW[2][128 * 256];   // 2 x 64 KB, double buffered
    const int tid  = threadIdx.x;
    const int lane = tid & 31, wave = tid >> 5;
    const int l15 = lane & 15, half = lane >> 4;
    const int mBlock = blockIdx.x * 128;       // spatial
    const int nBlock = blockIdx.y * 128;       // cout
    const int mWave = mBlock + (wave >> 1) * 32;
    const int nWaveL = (wave & 1) * 64;        // local within LDS tile

    // per-M-tile lane position (row in NHWC spatial space)
    int bI[2], hI[2], wI[2];
#pragma unroll
    for (int mt = 0; mt < 2; ++mt) {
        int row = mWave + mt * 16 + l15;
        bI[mt] = row >> 10; hI[mt] = (row >> 5) & 31; wI[mt] = row & 31;
    }

    f32x8 acc[2][4];
#pragma unroll
    for (int mt = 0; mt < 2; ++mt)
#pragma unroll
        for (int nt = 0; nt < 4; ++nt) acc[mt][nt] = zf8();

    // prologue: stage tap 0 into buffer 0
    if (HAVE_TDM) {
        if (wave == 0) {
            tdm_issue_2d(wt + (size_t)nBlock * KCONV, /*lds_off=*/0,
                         /*tile_w=*/CCH, /*tile_h=*/128, /*row_stride=*/KCONV);
            tdm_wait();
        }
    } else {
        for (int i = tid; i < 128 * 256 / 8; i += 256) {
            int idx = i * 8, row = idx >> 8, col = idx & 255;
            *(bf16x8*)(&ldsW[0][idx]) =
                *(const bf16x8*)(wt + (size_t)(nBlock + row) * KCONV + col);
        }
    }
    __syncthreads();   // publish buffer 0

    for (int tap = 0; tap < 9; ++tap) {
        const int cur = tap & 1;
        // async prefetch of next tap's panel into the other buffer; that
        // buffer was last read at tap-1, protected by the consumption barrier.
        if (HAVE_TDM && wave == 0 && tap + 1 < 9)
            tdm_issue_2d(wt + (size_t)nBlock * KCONV + (tap + 1) * CCH,
                         (unsigned)(sizeof(bf16) * 128 * 256 * ((tap + 1) & 1)),
                         CCH, 128, KCONV);

        const int dy = tap / 3 - 1, dx = tap % 3 - 1;
        const bf16* aPtr[2];
        bool valid[2];
#pragma unroll
        for (int mt = 0; mt < 2; ++mt) {
            int hh = hI[mt] + dy, ww = wI[mt] + dx;
            valid[mt] = ((unsigned)hh < 32u) && ((unsigned)ww < 32u);
            aPtr[mt] = act + (((size_t)(bI[mt] << 10) + (hh << 5) + ww) * CCH);
        }

        for (int kc = 0; kc < 8; ++kc) {       // 8 * 32 = 256 cin per tap
            bf16x16 fa[2];
#pragma unroll
            for (int mt = 0; mt < 2; ++mt)
                fa[mt] = valid[mt] ? load_a_frag(aPtr[mt] + kc * 32, half) : zb16();
#pragma unroll
            for (int nt = 0; nt < 4; ++nt) {
                bf16x16 fb = load_b_frag(
                    &ldsW[cur][(nWaveL + nt * 16 + l15) * 256 + kc * 32], half);
#pragma unroll
                for (int mt = 0; mt < 2; ++mt)
                    acc[mt][nt] = wmma_bf16(fa[mt], fb, acc[mt][nt]);
            }
        }
        __syncthreads();   // all waves done reading ldsW[cur]

        if (tap + 1 < 9) {
            if (HAVE_TDM) {
                if (wave == 0) tdm_wait();      // DMA of next panel complete
            } else {
                const int nxt = (tap + 1) & 1;
                for (int i = tid; i < 128 * 256 / 8; i += 256) {
                    int idx = i * 8, row = idx >> 8, col = idx & 255;
                    *(bf16x8*)(&ldsW[nxt][idx]) =
                        *(const bf16x8*)(wt + (size_t)(nBlock + row) * KCONV
                                            + (tap + 1) * CCH + col);
                }
            }
            __syncthreads();   // publish next panel
        }
    }
    // store f32 NHWC + bias
#pragma unroll
    for (int mt = 0; mt < 2; ++mt)
#pragma unroll
        for (int nt = 0; nt < 4; ++nt)
#pragma unroll
            for (int j = 0; j < 8; ++j) {
                int row = mWave + mt * 16 + j + 8 * half;
                int col = nBlock + nWaveL + nt * 16 + l15;
                out[(size_t)row * CCH + col] = acc[mt][nt][j] + bias[col];
            }
}

// ---------------------------------------------------------------------------
// Generic WMMA GEMM (K fixed = 256): D[M rows][N] = A[M][256] * Bw[N][256]^T
// Whole 128x256 B-panel (64 KB) staged ONCE via TDM; K loop is barrier-free.
// OUT_BF16: store bf16 [row][col].  FINAL: f32, + bias + residual(NHWC),
// store transposed to NCHW d_out (N must be 256).
// ---------------------------------------------------------------------------
template <bool OUT_BF16, bool FINAL>
__global__ __launch_bounds__(256) void gemm_wmma_kernel(
    const bf16* __restrict__ A, const bf16* __restrict__ Bw,
    const float* __restrict__ bias, const float* __restrict__ res,
    float* __restrict__ outF, bf16* __restrict__ outB, int N)
{
    __shared__ bf16 ldsW[128 * 256];   // 64 KB B-panel
    const int tid = threadIdx.x;
    const int lane = tid & 31, wave = tid >> 5;
    const int l15 = lane & 15, half = lane >> 4;
    const int mBlock = blockIdx.x * 128;
    const int nBlock = blockIdx.y * 128;
    const int mWave = mBlock + (wave >> 1) * 32;
    const int nWaveL = (wave & 1) * 64;
    const int K = 256;

    if (HAVE_TDM) {
        if (wave == 0) {
            tdm_issue_2d(Bw + (size_t)nBlock * K, 0, K, 128, K);
            tdm_wait();
        }
    } else {
        // panel is contiguous: rows stride == K
        const bf16* src = Bw + (size_t)nBlock * K;
        for (int i = tid; i < 128 * 256 / 8; i += 256)
            *(bf16x8*)(&ldsW[i * 8]) = *(const bf16x8*)(src + i * 8);
    }

    f32x8 acc[2][4];
#pragma unroll
    for (int mt = 0; mt < 2; ++mt)
#pragma unroll
        for (int nt = 0; nt < 4; ++nt) acc[mt][nt] = zf8();

    __syncthreads();   // publish B-panel

    for (int kb = 0; kb < K; kb += 32) {
        bf16x16 fa[2];
#pragma unroll
        for (int mt = 0; mt < 2; ++mt)
            fa[mt] = load_a_frag(A + (size_t)(mWave + mt * 16 + l15) * K + kb, half);
#pragma unroll
        for (int nt = 0; nt < 4; ++nt) {
            bf16x16 fb = load_b_frag(&ldsW[(nWaveL + nt * 16 + l15) * 256 + kb], half);
#pragma unroll
            for (int mt = 0; mt < 2; ++mt)
                acc[mt][nt] = wmma_bf16(fa[mt], fb, acc[mt][nt]);
        }
    }
#pragma unroll
    for (int mt = 0; mt < 2; ++mt)
#pragma unroll
        for (int nt = 0; nt < 4; ++nt)
#pragma unroll
            for (int j = 0; j < 8; ++j) {
                int row = mWave + mt * 16 + j + 8 * half;
                int col = nBlock + nWaveL + nt * 16 + l15;
                float v = acc[mt][nt][j];
                if (bias) v += bias[col];
                if (FINAL) {
                    v += res[(size_t)row * CCH + col];
                    int b = row >> 10, s = row & 1023;
                    outF[(((size_t)b * CCH + col) << 10) + s] = v;   // NCHW
                } else if (OUT_BF16) {
                    outB[(size_t)row * N + col] = (bf16)v;
                } else {
                    outF[(size_t)row * N + col] = v;
                }
            }
}

// ---------------------------------------------------------------------------
// Flash-style self attention (4 heads, hd=64, S=1024, scale=1/16).
// qkv bf16 [b*1024+s][768] (q at 0, k at 256); vt bf16 [b*256+c][1024].
// 8 independent waves per block; each wave owns 32 query rows.
// ---------------------------------------------------------------------------
__global__ __launch_bounds__(256) void flash_attn_kernel(
    const bf16* __restrict__ qkv, const bf16* __restrict__ vt,
    bf16* __restrict__ o)
{
    __shared__ bf16 ldsP[8][32 * 32];   // per-wave P tile (32 q x 32 keys)
    const int tid = threadIdx.x;
    const int lane = tid & 31, wave = tid >> 5;
    const int l15 = lane & 15, half = lane >> 4;
    const int b = blockIdx.x >> 2, head = blockIdx.x & 3;
    const int qBase = blockIdx.y * 256 + wave * 32;
    const float scale = 0.0625f;        // 1/sqrt(256)

    // preload q as A fragments: [mt][ks]  (K = hd = 64 -> 2 K-steps)
    bf16x16 faq[2][2];
#pragma unroll
    for (int mt = 0; mt < 2; ++mt)
#pragma unroll
        for (int ks = 0; ks < 2; ++ks) {
            const bf16* p = qkv + (size_t)((b << 10) + qBase + mt * 16 + l15) * 768
                                + head * 64 + ks * 32;
            faq[mt][ks] = load_a_frag(p, half);
        }

    f32x8 accO[2][4];
    float mrow[2][8], lrow[2][8];
#pragma unroll
    for (int mt = 0; mt < 2; ++mt) {
#pragma unroll
        for (int nt = 0; nt < 4; ++nt) accO[mt][nt] = zf8();
#pragma unroll
        for (int j = 0; j < 8; ++j) { mrow[mt][j] = -3.0e38f; lrow[mt][j] = 0.f; }
    }

    for (int kt = 0; kt < SPT; kt += 32) {
        // scores tile: 32 q x 32 keys
        f32x8 sc[2][2];
#pragma unroll
        for (int mt = 0; mt < 2; ++mt)
#pragma unroll
            for (int nt = 0; nt < 2; ++nt) sc[mt][nt] = zf8();
#pragma unroll
        for (int ks = 0; ks < 2; ++ks)
#pragma unroll
            for (int nt = 0; nt < 2; ++nt) {
                const bf16* kp = qkv + (size_t)((b << 10) + kt + nt * 16 + l15) * 768
                                     + 256 + head * 64 + ks * 32;
                bf16x16 fb = load_b_frag(kp, half);
#pragma unroll
                for (int mt = 0; mt < 2; ++mt)
                    sc[mt][nt] = wmma_bf16(faq[mt][ks], fb, sc[mt][nt]);
            }

        // online softmax + stage P into LDS (per-wave region)
#pragma unroll
        for (int mt = 0; mt < 2; ++mt) {
#pragma unroll
            for (int j = 0; j < 8; ++j) {
                float s0 = sc[mt][0][j] * scale;
                float s1 = sc[mt][1][j] * scale;
                float nm = fmaxf(s0, s1);
#pragma unroll
                for (int off = 8; off >= 1; off >>= 1)
                    nm = fmaxf(nm, __shfl_xor(nm, off, 32));
                float mold = mrow[mt][j];
                float mnew = fmaxf(mold, nm);
                float alpha = __expf(mold - mnew);
                float p0 = __expf(s0 - mnew);
                float p1 = __expf(s1 - mnew);
                float rs = p0 + p1;
#pragma unroll
                for (int off = 8; off >= 1; off >>= 1)
                    rs += __shfl_xor(rs, off, 32);
                mrow[mt][j] = mnew;
                lrow[mt][j] = lrow[mt][j] * alpha + rs;
#pragma unroll
                for (int nt = 0; nt < 4; ++nt) accO[mt][nt][j] *= alpha;
                int lr = mt * 16 + j + 8 * half;
                ldsP[wave][lr * 32 + l15]      = (bf16)p0;
                ldsP[wave][lr * 32 + 16 + l15] = (bf16)p1;
            }
        }

        // accO += P (32 q x 32 keys) * V^T (32 keys x 64 hd)
#pragma unroll
        for (int mt = 0; mt < 2; ++mt) {
            bf16x16 fap = load_a_frag(&ldsP[wave][(mt * 16 + l15) * 32], half);
#pragma unroll
            for (int nt = 0; nt < 4; ++nt) {
                const bf16* vp = vt + (size_t)(b * CCH + head * 64 + nt * 16 + l15) * SPT + kt;
                bf16x16 fbv = load_b_frag(vp, half);
                accO[mt][nt] = wmma_bf16(fap, fbv, accO[mt][nt]);
            }
        }
    }

    // epilogue: o /= l, store bf16 NHWC
#pragma unroll
    for (int mt = 0; mt < 2; ++mt)
#pragma unroll
        for (int nt = 0; nt < 4; ++nt)
#pragma unroll
            for (int j = 0; j < 8; ++j) {
                int row = (b << 10) + qBase + mt * 16 + j + 8 * half;
                int col = head * 64 + nt * 16 + l15;
                o[(size_t)row * CCH + col] = (bf16)(accO[mt][nt][j] / lrow[mt][j]);
            }
}

// ---------------------------------------------------------------------------
// Small helper kernels
// ---------------------------------------------------------------------------
__global__ void time_mlp_kernel(const float* __restrict__ t,
                                const float* __restrict__ w,
                                const float* __restrict__ bias,
                                float* __restrict__ emb)
{
    int b = blockIdx.x, oc = threadIdx.x;
    float acc = bias[oc];
    for (int c = 0; c < CCH; ++c) {
        float x = t[b * CCH + c];
        x = x / (1.f + __expf(-x));           // swish
        acc += x * w[oc * CCH + c];
    }
    emb[b * CCH + oc] = acc;
}

__global__ void add_emb_kernel(float* __restrict__ h, const float* __restrict__ emb)
{
    int i = blockIdx.x * 256 + threadIdx.x;   // NHWC index
    if (i >= MTOT * CCH) return;
    int b = i >> 18, c = i & 255;
    h[i] += emb[b * CCH + c];
}

__global__ void add_x_nchw_kernel(float* __restrict__ r, const float* __restrict__ x)
{
    int i = blockIdx.x * 256 + threadIdx.x;   // NHWC index
    if (i >= MTOT * CCH) return;
    int b = i >> 18, s = (i >> 8) & 1023, c = i & 255;
    r[i] += x[(((size_t)b * CCH + c) << 10) + s];
}

__global__ void conv_w_to_bf16_kernel(const float* __restrict__ w, bf16* __restrict__ wt)
{
    int i = blockIdx.x * 256 + threadIdx.x;   // wt index [cout][tap*256+cin]
    if (i >= CCH * KCONV) return;
    int cout = i / KCONV, rem = i % KCONV;
    int tap = rem >> 8, cin = rem & 255;
    wt[i] = (bf16)w[(size_t)cout * KCONV + cin * 9 + tap];
}

__global__ void f32_to_bf16_kernel(const float* __restrict__ in, bf16* __restrict__ out, int n)
{
    int i = blockIdx.x * 256 + threadIdx.x;
    if (i < n) out[i] = (bf16)in[i];
}

__global__ void transpose_v_kernel(const bf16* __restrict__ qkv, bf16* __restrict__ vt)
{
    int i = blockIdx.x * 256 + threadIdx.x;   // vt index [b*256+c][s]
    if (i >= BATCH * CCH * SPT) return;
    int b = i >> 18, c = (i >> 10) & 255, s = i & 1023;
    vt[i] = qkv[(((size_t)b << 10) + s) * 768 + 512 + c];
}

// ---------------------------------------------------------------------------
// Workspace byte offsets
// ---------------------------------------------------------------------------
#define WS_A1    ((size_t)0)            // act1 bf16 NHWC         4 MiB
#define WS_H1    ((size_t)4  << 20)     // conv1 out f32 NHWC     8 MiB
#define WS_A2    ((size_t)12 << 20)     // act2 bf16 NHWC         4 MiB
#define WS_H2    ((size_t)16 << 20)     // conv2 out / r f32 NHWC 8 MiB
#define WS_NB    ((size_t)24 << 20)     // attn-GN out bf16       4 MiB
#define WS_QKV   ((size_t)28 << 20)     // qkv bf16 [8192][768]  12 MiB
#define WS_VT    ((size_t)40 << 20)     // v^T bf16 [2048][1024]  4 MiB
#define WS_AO    ((size_t)44 << 20)     // attn out bf16 NHWC     4 MiB
#define WS_W1T   ((size_t)48 << 20)     // conv1 w bf16           1.2 MiB
#define WS_W2T   ((size_t)50 << 20)     // conv2 w bf16           1.2 MiB
#define WS_WQKV  ((size_t)52 << 20)     // qkv w bf16             384 KiB
#define WS_WOUT  ((size_t)53 << 20)     // out w bf16             128 KiB
#define WS_EMB   ((size_t)54 << 20)     // emb f32                8 KiB

extern "C" void kernel_launch(void* const* d_in, const int* in_sizes, int n_in,
                              void* d_out, int out_size, void* d_ws, size_t ws_size,
                              hipStream_t stream)
{
    const float* x        = (const float*)d_in[0];
    const float* time_emb = (const float*)d_in[1];
    const float* gn1_s    = (const float*)d_in[2];
    const float* gn1_b    = (const float*)d_in[3];
    const float* conv1_w  = (const float*)d_in[4];
    const float* conv1_b  = (const float*)d_in[5];
    const float* mlp_w    = (const float*)d_in[6];
    const float* mlp_b    = (const float*)d_in[7];
    const float* gn2_s    = (const float*)d_in[8];
    const float* gn2_b    = (const float*)d_in[9];
    const float* conv2_w  = (const float*)d_in[10];
    const float* conv2_b  = (const float*)d_in[11];
    const float* agn_s    = (const float*)d_in[12];
    const float* agn_b    = (const float*)d_in[13];
    const float* qkv_w    = (const float*)d_in[14];
    const float* out_w    = (const float*)d_in[15];
    const float* out_b    = (const float*)d_in[16];

    char* ws = (char*)d_ws;
    bf16*  a1   = (bf16*)(ws + WS_A1);
    float* h1   = (float*)(ws + WS_H1);
    bf16*  a2   = (bf16*)(ws + WS_A2);
    float* r    = (float*)(ws + WS_H2);
    bf16*  nb   = (bf16*)(ws + WS_NB);
    bf16*  qkv  = (bf16*)(ws + WS_QKV);
    bf16*  vt   = (bf16*)(ws + WS_VT);
    bf16*  ao   = (bf16*)(ws + WS_AO);
    bf16*  w1t  = (bf16*)(ws + WS_W1T);
    bf16*  w2t  = (bf16*)(ws + WS_W2T);
    bf16*  wq   = (bf16*)(ws + WS_WQKV);
    bf16*  wo   = (bf16*)(ws + WS_WOUT);
    float* emb  = (float*)(ws + WS_EMB);
    float* outp = (float*)d_out;

    // --- weight conversions (f32 -> bf16, conv weights re-laid tap-major)
    conv_w_to_bf16_kernel<<<(CCH * KCONV + 255) / 256, 256, 0, stream>>>(conv1_w, w1t);
    conv_w_to_bf16_kernel<<<(CCH * KCONV + 255) / 256, 256, 0, stream>>>(conv2_w, w2t);
    f32_to_bf16_kernel<<<(3 * CCH * CCH + 255) / 256, 256, 0, stream>>>(qkv_w, wq, 3 * CCH * CCH);
    f32_to_bf16_kernel<<<(CCH * CCH + 255) / 256, 256, 0, stream>>>(out_w, wo, CCH * CCH);

    // --- ResnetBlock
    groupnorm_kernel<true, true><<<256, 256, 0, stream>>>(x, gn1_s, gn1_b, a1);
    conv3x3_wmma_kernel<<<dim3(64, 2), 256, 0, stream>>>(a1, w1t, conv1_b, h1);
    time_mlp_kernel<<<BATCH, 256, 0, stream>>>(time_emb, mlp_w, mlp_b, emb);
    add_emb_kernel<<<(MTOT * CCH + 255) / 256, 256, 0, stream>>>(h1, emb);
    groupnorm_kernel<false, true><<<256, 256, 0, stream>>>(h1, gn2_s, gn2_b, a2);
    conv3x3_wmma_kernel<<<dim3(64, 2), 256, 0, stream>>>(a2, w2t, conv2_b, r);
    add_x_nchw_kernel<<<(MTOT * CCH + 255) / 256, 256, 0, stream>>>(r, x);   // r = h + x

    // --- SelfAttention
    groupnorm_kernel<false, false><<<256, 256, 0, stream>>>(r, agn_s, agn_b, nb);
    gemm_wmma_kernel<true, false><<<dim3(64, 6), 256, 0, stream>>>(
        nb, wq, nullptr, nullptr, nullptr, qkv, 3 * CCH);
    transpose_v_kernel<<<(BATCH * CCH * SPT + 255) / 256, 256, 0, stream>>>(qkv, vt);
    flash_attn_kernel<<<dim3(32, 4), 256, 0, stream>>>(qkv, vt, ao);

    // --- out projection + bias + residual, store NCHW into d_out
    gemm_wmma_kernel<false, true><<<dim3(64, 2), 256, 0, stream>>>(
        ao, wo, out_b, r, outp, nullptr, CCH);
}
